// VectorQuantizerAE_70763881168915
// MI455X (gfx1250) — matrix operand order repaired and checked by reading, not compiled, not run on hardware
//
#include <hip/hip_runtime.h>

typedef __attribute__((ext_vector_type(16))) _Float16 v16h;
typedef __attribute__((ext_vector_type(8)))  _Float16 v8h;
typedef __attribute__((ext_vector_type(8)))  float    v8f;

#define NTOT 65536
#define FD   768
#define WID  256
#define ED   6
#define NEB  1024

static constexpr size_t OFF_REC    = (size_t)NTOT * FD;        // 50331648
static constexpr size_t OFF_COMMIT = OFF_REC + 1;
static constexpr size_t OFF_KL     = OFF_REC + 2;
static constexpr size_t OFF_LB     = OFF_REC + 3;
static constexpr size_t OFF_CB     = OFF_REC + 4;
static constexpr size_t OFF_H      = OFF_CB + (size_t)NEB * ED;
static constexpr size_t OFF_PERP   = OFF_H + (size_t)NTOT * ED;

// ---------------- device helpers ----------------

__device__ __forceinline__ float siluf(float x) { return x / (1.f + __expf(-x)); }

// A-operand (16x32 f16) gather from LDS, per documented CDNA5 lane layout:
// lanes 0-15: row=lane, K in {k*32+0..7, k*32+16..23}; lanes 16-31: row=lane-16, K+8.
__device__ __forceinline__ v16h ldsA(const _Float16* x, int stride, int k, int lane) {
    int rA   = lane & 15;
    int base = k * 32 + ((lane >> 4) & 1) * 8;
    const _Float16* p = x + rA * stride + base;
    v8h lo = *(const v8h*)(p);
    v8h hi = *(const v8h*)(p + 16);
    v16h a;
#pragma unroll
    for (int i = 0; i < 8; ++i) { a[i] = lo[i]; a[8 + i] = hi[i]; }
    return a;
}

// B-operand (32x16 f16): weights pre-swizzled so each lane reads 16 contiguous halves.
__device__ __forceinline__ v16h glbB(const _Float16* w, int tile, int KS, int k, int lane) {
    return *(const v16h*)(w + ((size_t)((tile * KS + k) * 32 + lane) << 4));
}

// ---------------- prep kernels ----------------

__global__ void initK(unsigned* cnt) {
    int i = blockIdx.x * 256 + threadIdx.x;
    if (i < NEB) cnt[i] = 0u;
}

__global__ void cbNormK(const float* __restrict__ cb, float* __restrict__ cbN,
                        float* __restrict__ outCb) {
    int i = blockIdx.x * 256 + threadIdx.x;
    if (i < NEB) {
        float v[6]; float s = 0.f;
#pragma unroll
        for (int e = 0; e < 6; ++e) { v[e] = cb[i * 6 + e]; s += v[e] * v[e]; }
        float n = fmaxf(sqrtf(s), 1e-12f);
#pragma unroll
        for (int e = 0; e < 6; ++e) { float q = v[e] / n; cbN[i * 6 + e] = q; outCb[i * 6 + e] = q; }
    }
}

// Swizzle f32 weight [K x Nfull] into WMMA-B f16 lane-major tiles.
__global__ void swizzleB(const float* __restrict__ W, _Float16* __restrict__ dst,
                         int KS, int NT, int Nfull) {
    int total = KS * NT * 512;
    for (int o = blockIdx.x * blockDim.x + threadIdx.x; o < total; o += gridDim.x * blockDim.x) {
        int j  = o & 15;
        int l  = (o >> 4) & 31;
        int kn = o >> 9;             // n*KS + k
        int k  = kn % KS;
        int n  = kn / KS;
        int kk  = k * 32 + ((l >> 4) & 1) * 8 + (j & 7) + (j >> 3) * 16;
        int col = n * 16 + (l & 15);
        dst[o] = (_Float16)W[(size_t)kk * Nfull + col];
    }
}

// ---------------- encoder: LN0 -> W1(WMMA) -> SiLU -> LN1 -> W2(WMMA) -> SiLU -> LN2 -> W3 -> normalize ----------------

__global__ __launch_bounds__(256) void encoderK(
    const float* __restrict__ z,
    const float* __restrict__ g0, const float* __restrict__ c0,
    const _Float16* __restrict__ w1sw, const float* __restrict__ b1,
    const float* __restrict__ g1, const float* __restrict__ c1,
    const _Float16* __restrict__ w2sw, const float* __restrict__ b2,
    const float* __restrict__ g2, const float* __restrict__ c2,
    const float* __restrict__ W3, const float* __restrict__ b3,
    float* __restrict__ hOut)
{
    __shared__ alignas(32) unsigned char smem[43008];
    _Float16* xln  = (_Float16*)smem;            // phase1: [16][768] f16
    _Float16* xln2 = (_Float16*)smem;            // phase2: [16][256] f16 (reuse)
    float* ln2f = (float*)(smem + 8192);         // [16][256] f32
    float* hbuf = (float*)(smem + 24576);        // [16][256] f32
    float* red  = (float*)(smem + 40960);        // [256]
    float* red2 = (float*)(smem + 41984);        // [256]

    const int tid = threadIdx.x;
    const int lane = tid & 31, wave = tid >> 5;
    const int row = tid >> 4,  sub  = tid & 15;
    const size_t gBase = (size_t)blockIdx.x * 16;

    // ---- LN0 over 768 + f16 convert into LDS ----
    {
        float v[48];
        const float* zr = z + (gBase + row) * FD;
        float s = 0.f, s2 = 0.f;
#pragma unroll
        for (int i = 0; i < 48; ++i) { float x = zr[sub + 16 * i]; v[i] = x; s += x; s2 += x * x; }
        red[tid] = s; red2[tid] = s2;
        __syncthreads();
        float ts = 0.f, ts2 = 0.f;
        for (int j = 0; j < 16; ++j) { ts += red[row * 16 + j]; ts2 += red2[row * 16 + j]; }
        float mean = ts * (1.f / 768.f);
        float rstd = rsqrtf(ts2 * (1.f / 768.f) - mean * mean + 1e-5f);
#pragma unroll
        for (int i = 0; i < 48; ++i) {
            int cc = sub + 16 * i;
            xln[row * 768 + cc] = (_Float16)((v[i] - mean) * rstd * g0[cc] + c0[cc]);
        }
        __syncthreads();
    }

    // ---- GEMM1: [16x768] @ [768x256] via WMMA, +b1, SiLU -> hbuf ----
    for (int t2 = 0; t2 < 2; ++t2) {
        const int tile = wave + t2 * 8;
        v8f acc = {};
        for (int k = 0; k < 24; ++k) {
            v16h a = ldsA(xln, 768, k, lane);
            v16h b = glbB(w1sw, tile, 24, k, lane);
            __builtin_prefetch((const void*)(w1sw + ((size_t)((tile * 24 + ((k + 1) % 24)) * 32 + lane) << 4)), 0, 1);
            acc = __builtin_amdgcn_wmma_f32_16x16x32_f16(false, a, false, b, (short)0, acc, false, false);
        }
        const int col = tile * 16 + (lane & 15);
#pragma unroll
        for (int r = 0; r < 8; ++r) {
            int M = r + ((lane >> 4) & 1) * 8;
            hbuf[M * 256 + col] = siluf(acc[r] + b1[col]);
        }
    }
    __syncthreads();

    // ---- LN1 over 256 -> xln2 f16 ----
    {
        float u[16]; float s = 0.f, s2 = 0.f;
#pragma unroll
        for (int i = 0; i < 16; ++i) { float x = hbuf[row * 256 + sub + 16 * i]; u[i] = x; s += x; s2 += x * x; }
        red[tid] = s; red2[tid] = s2;
        __syncthreads();
        float ts = 0.f, ts2 = 0.f;
        for (int j = 0; j < 16; ++j) { ts += red[row * 16 + j]; ts2 += red2[row * 16 + j]; }
        float mean = ts * (1.f / 256.f);
        float rstd = rsqrtf(ts2 * (1.f / 256.f) - mean * mean + 1e-5f);
#pragma unroll
        for (int i = 0; i < 16; ++i) {
            int cc = sub + 16 * i;
            xln2[row * 256 + cc] = (_Float16)((u[i] - mean) * rstd * g1[cc] + c1[cc]);
        }
        __syncthreads();
    }

    // ---- GEMM2: [16x256] @ [256x256] via WMMA, +b2, SiLU -> hbuf ----
    for (int t2 = 0; t2 < 2; ++t2) {
        const int tile = wave + t2 * 8;
        v8f acc = {};
#pragma unroll
        for (int k = 0; k < 8; ++k) {
            v16h a = ldsA(xln2, 256, k, lane);
            v16h b = glbB(w2sw, tile, 8, k, lane);
            acc = __builtin_amdgcn_wmma_f32_16x16x32_f16(false, a, false, b, (short)0, acc, false, false);
        }
        const int col = tile * 16 + (lane & 15);
#pragma unroll
        for (int r = 0; r < 8; ++r) {
            int M = r + ((lane >> 4) & 1) * 8;
            hbuf[M * 256 + col] = siluf(acc[r] + b2[col]);
        }
    }
    __syncthreads();

    // ---- LN2 -> ln2f (f32) ----
    {
        float u[16]; float s = 0.f, s2 = 0.f;
#pragma unroll
        for (int i = 0; i < 16; ++i) { float x = hbuf[row * 256 + sub + 16 * i]; u[i] = x; s += x; s2 += x * x; }
        red[tid] = s; red2[tid] = s2;
        __syncthreads();
        float ts = 0.f, ts2 = 0.f;
        for (int j = 0; j < 16; ++j) { ts += red[row * 16 + j]; ts2 += red2[row * 16 + j]; }
        float mean = ts * (1.f / 256.f);
        float rstd = rsqrtf(ts2 * (1.f / 256.f) - mean * mean + 1e-5f);
#pragma unroll
        for (int i = 0; i < 16; ++i) {
            int cc = sub + 16 * i;
            ln2f[row * 256 + cc] = (u[i] - mean) * rstd * g2[cc] + c2[cc];
        }
        __syncthreads();
    }

    // ---- W3 (256->6) + L2-normalize -> hOut ----
    if (tid < 96) {
        int r3 = tid / 6, e = tid - r3 * 6;
        float a = b3[e];
        for (int k2 = 0; k2 < 256; ++k2) a += ln2f[r3 * 256 + k2] * W3[k2 * 6 + e];
        red[tid] = a;
    }
    __syncthreads();
    if (tid < 16) {
        float hv[6]; float sn = 0.f;
#pragma unroll
        for (int e = 0; e < 6; ++e) { hv[e] = red[tid * 6 + e]; sn += hv[e] * hv[e]; }
        float nn = fmaxf(sqrtf(sn), 1e-12f);
#pragma unroll
        for (int e = 0; e < 6; ++e) hOut[(gBase + tid) * ED + e] = hv[e] / nn;
    }
}

// ---------------- VQ: cosine sims, argmax, softmax stats ----------------

__global__ __launch_bounds__(256) void vqK(
    const float* __restrict__ hIn, const float* __restrict__ cbN,
    float* __restrict__ pPart, unsigned* __restrict__ cnt,
    float* __restrict__ commitPart, unsigned* __restrict__ idxOut,
    int rowsPerBlock)
{
    __shared__ float    cbs[NEB * ED];
    __shared__ float    pL[NEB];
    __shared__ unsigned cL[NEB];
    __shared__ float    rM[256];
    __shared__ int      rI[256];
    __shared__ float    rS[256];
    __shared__ float    rowInfo[2];

    const int tid = threadIdx.x;
    for (int i = tid; i < NEB * ED; i += 256) cbs[i] = cbN[i];
#pragma unroll
    for (int s = 0; s < 4; ++s) { pL[tid * 4 + s] = 0.f; cL[tid * 4 + s] = 0u; }
    float commitL = 0.f;
    __syncthreads();

    const size_t rowBase = (size_t)blockIdx.x * rowsPerBlock;
    for (int rr = 0; rr < rowsPerBlock; ++rr) {
        size_t rrow = rowBase + rr;
        float h0 = hIn[rrow * 6 + 0], h1 = hIn[rrow * 6 + 1], h2 = hIn[rrow * 6 + 2],
              h3 = hIn[rrow * 6 + 3], h4 = hIn[rrow * 6 + 4], h5 = hIn[rrow * 6 + 5];
        float ex[4]; float tmax = -1e30f; int tix = 0; float tsum = 0.f;
#pragma unroll
        for (int s = 0; s < 4; ++s) {
            int j = tid * 4 + s;
            const float* cp = cbs + j * 6;
            float d = h0 * cp[0] + h1 * cp[1] + h2 * cp[2] + h3 * cp[3] + h4 * cp[4] + h5 * cp[5];
            float e = __expf(d); ex[s] = e; tsum += e;
            if (d > tmax) { tmax = d; tix = j; }
        }
        rM[tid] = tmax; rI[tid] = tix; rS[tid] = tsum;
        __syncthreads();
        if (tid == 0) {   // first-max tie-break matches jnp.argmax (ascending j ownership)
            float bm = -1e30f; int bi = 0; float ss = 0.f;
            for (int t = 0; t < 256; ++t) { ss += rS[t]; if (rM[t] > bm) { bm = rM[t]; bi = rI[t]; } }
            rowInfo[0] = bm; rowInfo[1] = ss;
            cL[bi] += 1u; commitL += bm; idxOut[rrow] = (unsigned)bi;
        }
        __syncthreads();
        float inv = 1.f / rowInfo[1];
#pragma unroll
        for (int s = 0; s < 4; ++s) pL[tid * 4 + s] += ex[s] * inv;
        __syncthreads();
    }
#pragma unroll
    for (int s = 0; s < 4; ++s) {
        int j = tid * 4 + s;
        pPart[(size_t)blockIdx.x * NEB + j] = pL[j];
        if (cL[j]) atomicAdd(&cnt[j], cL[j]);       // integer: exact & deterministic
    }
    if (tid == 0) commitPart[blockIdx.x] = commitL;
}

// ---------------- decoder: gather z_q -> LN0 -> W1(6->256) -> SiLU -> LN1 -> W2(WMMA) -> SiLU -> LN2 -> W3(WMMA) + rec partials ----------------

__global__ __launch_bounds__(256) void decoderK(
    const unsigned* __restrict__ qidx, const float* __restrict__ cbN,
    const float* __restrict__ g0, const float* __restrict__ c0,
    const float* __restrict__ W1, const float* __restrict__ b1,
    const float* __restrict__ g1, const float* __restrict__ c1,
    const _Float16* __restrict__ w2sw, const float* __restrict__ b2,
    const float* __restrict__ g2, const float* __restrict__ c2,
    const _Float16* __restrict__ w3sw, const float* __restrict__ b3,
    const float* __restrict__ z, float* __restrict__ zhat,
    float* __restrict__ s1Part, float* __restrict__ s2Part)
{
    __shared__ alignas(32) unsigned char smem[35072];
    _Float16* xln  = (_Float16*)smem;             // [16][256]
    _Float16* xln2 = (_Float16*)(smem + 8192);    // [16][256]
    float* hbuf   = (float*)(smem + 16384);       // [16][256]
    float* red    = (float*)(smem + 32768);
    float* red2   = (float*)(smem + 33792);
    float* rowAcc = (float*)(smem + 34816);       // [16][4]

    const int tid = threadIdx.x, lane = tid & 31, wave = tid >> 5;
    const int row = tid >> 4, sub = tid & 15;
    const size_t gBase = (size_t)blockIdx.x * 16;

    // ---- gather z_q, LN0(6), Linear 6->256, SiLU, LN1 -> xln ----
    {
        unsigned qi = qidx[gBase + row];
        float cv[6]; float s = 0.f;
#pragma unroll
        for (int e = 0; e < 6; ++e) { cv[e] = cbN[qi * 6 + e]; s += cv[e]; }
        float mean = s * (1.f / 6.f);
        float s2 = 0.f;
#pragma unroll
        for (int e = 0; e < 6; ++e) { float d = cv[e] - mean; s2 += d * d; }
        float rstd = rsqrtf(s2 * (1.f / 6.f) + 1e-5f);
        float xq[6];
#pragma unroll
        for (int e = 0; e < 6; ++e) xq[e] = (cv[e] - mean) * rstd * g0[e] + c0[e];

        float yv[16]; float ys = 0.f, ys2 = 0.f;
#pragma unroll
        for (int i = 0; i < 16; ++i) {
            int cc = sub + 16 * i;
            float a = b1[cc];
#pragma unroll
            for (int e = 0; e < 6; ++e) a += xq[e] * W1[e * 256 + cc];
            a = siluf(a);
            yv[i] = a; ys += a; ys2 += a * a;
        }
        red[tid] = ys; red2[tid] = ys2;
        __syncthreads();
        float ts = 0.f, ts2 = 0.f;
        for (int j = 0; j < 16; ++j) { ts += red[row * 16 + j]; ts2 += red2[row * 16 + j]; }
        float m2 = ts * (1.f / 256.f);
        float r2 = rsqrtf(ts2 * (1.f / 256.f) - m2 * m2 + 1e-5f);
#pragma unroll
        for (int i = 0; i < 16; ++i) {
            int cc = sub + 16 * i;
            xln[row * 256 + cc] = (_Float16)((yv[i] - m2) * r2 * g1[cc] + c1[cc]);
        }
        __syncthreads();
    }

    // ---- GEMM W2 (WMMA) + SiLU -> hbuf ----
    for (int t2 = 0; t2 < 2; ++t2) {
        int tile = wave + t2 * 8;
        v8f acc = {};
#pragma unroll
        for (int k = 0; k < 8; ++k) {
            v16h a = ldsA(xln, 256, k, lane);
            v16h b = glbB(w2sw, tile, 8, k, lane);
            acc = __builtin_amdgcn_wmma_f32_16x16x32_f16(false, a, false, b, (short)0, acc, false, false);
        }
        int col = tile * 16 + (lane & 15);
#pragma unroll
        for (int r = 0; r < 8; ++r) {
            int M = r + ((lane >> 4) & 1) * 8;
            hbuf[M * 256 + col] = siluf(acc[r] + b2[col]);
        }
    }
    __syncthreads();

    // ---- LN2 -> xln2 f16 ----
    {
        float u[16]; float s = 0.f, s2 = 0.f;
#pragma unroll
        for (int i = 0; i < 16; ++i) { float x = hbuf[row * 256 + sub + 16 * i]; u[i] = x; s += x; s2 += x * x; }
        red[tid] = s; red2[tid] = s2;
        __syncthreads();
        float ts = 0.f, ts2 = 0.f;
        for (int j = 0; j < 16; ++j) { ts += red[row * 16 + j]; ts2 += red2[row * 16 + j]; }
        float mean = ts * (1.f / 256.f);
        float rstd = rsqrtf(ts2 * (1.f / 256.f) - mean * mean + 1e-5f);
#pragma unroll
        for (int i = 0; i < 16; ++i) {
            int cc = sub + 16 * i;
            xln2[row * 256 + cc] = (_Float16)((u[i] - mean) * rstd * g2[cc] + c2[cc]);
        }
        if (tid < 64) rowAcc[tid] = 0.f;
        __syncthreads();
    }

    // ---- GEMM W3: [16x256] @ [256x768] (WMMA) + bias -> zhat; rec-loss partials ----
    {
        float dz[8] = {0}, nh[8] = {0}, nz[8] = {0}, sq[8] = {0};
        for (int t = 0; t < 6; ++t) {
            int tile = wave * 6 + t;
            v8f acc = {};
#pragma unroll
            for (int k = 0; k < 8; ++k) {
                v16h a = ldsA(xln2, 256, k, lane);
                v16h b = glbB(w3sw, tile, 8, k, lane);
                acc = __builtin_amdgcn_wmma_f32_16x16x32_f16(false, a, false, b, (short)0, acc, false, false);
            }
            int col = tile * 16 + (lane & 15);
#pragma unroll
            for (int r = 0; r < 8; ++r) {
                int M = r + ((lane >> 4) & 1) * 8;
                float val = acc[r] + b3[col];
                size_t gi = (gBase + M) * (size_t)FD + col;
                zhat[gi] = val;
                float zv = z[gi];
                dz[r] += val * zv; nh[r] += val * val; nz[r] += zv * zv;
                float df = val - zv; sq[r] += df * df;
            }
        }
#pragma unroll
        for (int r = 0; r < 8; ++r) {
            int M = r + ((lane >> 4) & 1) * 8;
            atomicAdd(&rowAcc[M * 4 + 0], dz[r]);
            atomicAdd(&rowAcc[M * 4 + 1], nh[r]);
            atomicAdd(&rowAcc[M * 4 + 2], nz[r]);
            atomicAdd(&rowAcc[M * 4 + 3], sq[r]);
        }
    }
    __syncthreads();
    if (tid < 16) {
        float dot = rowAcc[tid * 4 + 0], a = rowAcc[tid * 4 + 1];
        float b   = rowAcc[tid * 4 + 2], sd = rowAcc[tid * 4 + 3];
        red[tid]  = dot / (fmaxf(sqrtf(a), 1e-6f) * fmaxf(sqrtf(b), 1e-6f));
        red2[tid] = sd;
    }
    __syncthreads();
    if (tid == 0) {
        float S1 = 0.f, S2 = 0.f;
        for (int j = 0; j < 16; ++j) { S1 += red[j]; S2 += red2[j]; }
        s1Part[blockIdx.x] = S1; s2Part[blockIdx.x] = S2;
    }
}

// ---------------- final scalar reduction (deterministic fixed-order) ----------------

__global__ __launch_bounds__(256) void finalK(
    const float* __restrict__ pPart, const unsigned* __restrict__ cnt,
    const float* __restrict__ commitPart,
    const float* __restrict__ s1Part, const float* __restrict__ s2Part,
    float* __restrict__ out)
{
    __shared__ float r1[256], r2[256], r3[256], r4[256], r5[256];
    const int tid = threadIdx.x;
    float kl = 0.f, lb = 0.f, pp = 0.f;
#pragma unroll
    for (int s = 0; s < 4; ++s) {
        int j = tid * 4 + s;
        float acc = 0.f;
        for (int b = 0; b < 512; ++b) acc += pPart[(size_t)b * NEB + j];
        float p = acc * (1.f / 65536.f);
        kl += p * (__logf(p) + 6.9314718055994531f);     // log p - log(1/1024)
        float e = (float)cnt[j] * (1.f / 65536.f);
        lb += e * p;
        pp += e * __logf(e + 1e-6f);
    }
    float s1 = 0.f, s2 = 0.f;
    for (int i = 0; i < 16; ++i) { int b = tid * 16 + i; s1 += s1Part[b]; s2 += s2Part[b]; }
    r1[tid] = kl; r2[tid] = lb; r3[tid] = pp; r4[tid] = s1; r5[tid] = s2;
    __syncthreads();
    if (tid == 0) {
        float K = 0.f, L = 0.f, P = 0.f, S1 = 0.f, S2 = 0.f, C = 0.f;
        for (int t = 0; t < 256; ++t) { K += r1[t]; L += r2[t]; P += r3[t]; S1 += r4[t]; S2 += r5[t]; }
        for (int b = 0; b < 512; ++b) C += commitPart[b];
        out[OFF_KL]     = K;
        out[OFF_LB]     = L;
        out[OFF_PERP]   = __expf(-P);
        out[OFF_COMMIT] = 1.25f * (1.f - C * (1.f / 65536.f));   // (1+BETA)*(1-mean(d_max))
        out[OFF_REC]    = 1.f - S1 * (1.f / 65536.f)
                          + 0.001f * (S2 * (1.f / (65536.f * 768.f)));
    }
}

// ---------------- host launch ----------------

extern "C" void kernel_launch(void* const* d_in, const int* in_sizes, int n_in,
                              void* d_out, int out_size, void* d_ws, size_t ws_size,
                              hipStream_t stream) {
    (void)in_sizes; (void)n_in; (void)out_size; (void)ws_size;

    const float* z     = (const float*)d_in[0];
    // encoder params
    const float* e_g0 = (const float*)d_in[1];
    const float* e_c0 = (const float*)d_in[2];
    const float* e_W1 = (const float*)d_in[3];
    const float* e_b1 = (const float*)d_in[4];
    const float* e_g1 = (const float*)d_in[5];
    const float* e_c1 = (const float*)d_in[6];
    const float* e_W2 = (const float*)d_in[7];
    const float* e_b2 = (const float*)d_in[8];
    const float* e_g2 = (const float*)d_in[9];
    const float* e_c2 = (const float*)d_in[10];
    const float* e_W3 = (const float*)d_in[11];
    const float* e_b3 = (const float*)d_in[12];
    // decoder params
    const float* d_g0 = (const float*)d_in[13];
    const float* d_c0 = (const float*)d_in[14];
    const float* d_W1 = (const float*)d_in[15];
    const float* d_b1 = (const float*)d_in[16];
    const float* d_g1 = (const float*)d_in[17];
    const float* d_c1 = (const float*)d_in[18];
    const float* d_W2 = (const float*)d_in[19];
    const float* d_b2 = (const float*)d_in[20];
    const float* d_g2 = (const float*)d_in[21];
    const float* d_c2 = (const float*)d_in[22];
    const float* d_W3 = (const float*)d_in[23];
    const float* d_b3 = (const float*)d_in[24];
    const float* codebook = (const float*)d_in[25];

    float* out = (float*)d_out;
    char*  ws  = (char*)d_ws;

    _Float16* w1sw   = (_Float16*)(ws + 0);          // 393216 B
    _Float16* w2sw   = (_Float16*)(ws + 393216);     // 131072 B
    _Float16* w2dsw  = (_Float16*)(ws + 524288);     // 131072 B
    _Float16* w3dsw  = (_Float16*)(ws + 655360);     // 393216 B
    float*    cbN    = (float*)(ws + 1048576);       //  24576 B
    unsigned* idx    = (unsigned*)(ws + 1073152);    // 262144 B
    unsigned* cnt    = (unsigned*)(ws + 1335296);    //   4096 B
    float*    commitPart = (float*)(ws + 1339392);   //   2048 B (512)
    float*    s1Part = (float*)(ws + 1341440);       //  16384 B (4096)
    float*    s2Part = (float*)(ws + 1357824);       //  16384 B (4096)
    float*    pPart  = (float*)(ws + 1374208);       // 2097152 B (512*1024)

    initK<<<4, 256, 0, stream>>>(cnt);
    cbNormK<<<4, 256, 0, stream>>>(codebook, cbN, out + OFF_CB);
    swizzleB<<<192, 256, 0, stream>>>(e_W1, w1sw, 24, 16, 256);
    swizzleB<<<64,  256, 0, stream>>>(e_W2, w2sw,  8, 16, 256);
    swizzleB<<<64,  256, 0, stream>>>(d_W2, w2dsw, 8, 16, 256);
    swizzleB<<<192, 256, 0, stream>>>(d_W3, w3dsw, 8, 48, 768);

    encoderK<<<NTOT / 16, 256, 0, stream>>>(
        z, e_g0, e_c0, w1sw, e_b1, e_g1, e_c1, w2sw, e_b2, e_g2, e_c2,
        e_W3, e_b3, out + OFF_H);

    vqK<<<512, 256, 0, stream>>>(out + OFF_H, cbN, pPart, cnt, commitPart, idx,
                                 NTOT / 512);

    decoderK<<<NTOT / 16, 256, 0, stream>>>(
        idx, cbN, d_g0, d_c0, d_W1, d_b1, d_g1, d_c1, w2dsw, d_b2,
        d_g2, d_c2, w3dsw, d_b3, z, out, s1Part, s2Part);

    finalK<<<1, 256, 0, stream>>>(pPart, cnt, commitPart, s1Part, s2Part, out);
}
